// LinearAttention_57578331570782
// MI455X (gfx1250) — matrix operand order repaired and checked by reading, not compiled, not run on hardware
//
#include <hip/hip_runtime.h>
#include <math.h>

// ---------------- problem constants ----------------
#define BB      4
#define SS      2048
#define SPAD    (SS + 2)      // padded rows per batch for causal conv
#define FF      512
#define INTER   1024
#define NEXP    16
#define TOK     8192          // BB*SS
#define CAP     512           // TOK/NEXP
#define N3I     3072          // 3*INTER

// ---------------- WMMA / vector types ----------------
typedef __attribute__((ext_vector_type(16))) __bf16    v16bf;
typedef __attribute__((ext_vector_type(8)))  float     v8f;
typedef __attribute__((ext_vector_type(4)))  unsigned  v4u;   // POD 16B vector

__device__ __forceinline__ unsigned short f2bf_bits(float f) {
    unsigned u = __float_as_uint(f);
    u += 0x7FFFu + ((u >> 16) & 1u);           // round-to-nearest-even
    return (unsigned short)(u >> 16);
}
__device__ __forceinline__ __bf16 f2bf(float f) {
    unsigned short h = f2bf_bits(f);
    return __builtin_bit_cast(__bf16, h);
}

struct BPack { v4u lo, hi; };                  // 32 bytes

// force global (address_space 1) b128 loads: avoids flat_load (which also
// ticks DScnt and takes the LDS path) in the GEMM hot loop
typedef __attribute__((address_space(1))) const v4u g_v4u;

__device__ __forceinline__ v16bf load_frag(const __bf16* p0, const __bf16* p1) {
    BPack t;
    t.lo = *(g_v4u*)(unsigned long long)p0;    // 16B aligned, global AS
    t.hi = *(g_v4u*)(unsigned long long)p1;
    return __builtin_bit_cast(v16bf, t);
}

// =====================================================================
// 0a) transpose+convert inp [B,F,S] f32 -> xbf [T,F] bf16 (LDS tile)
// =====================================================================
__global__ __launch_bounds__(256) void tcvt_kernel(
    const float* __restrict__ inp, __bf16* __restrict__ xbf)
{
    __shared__ float tile[32][33];
    const int tid = threadIdx.x;
    const int s0 = blockIdx.x * 32;            // global token tile base
    const int f0 = blockIdx.y * 32;
    const int b = s0 >> 11, sb = s0 & (SS - 1);
    #pragma unroll
    for (int r = 0; r < 4; ++r) {
        const int fl = (tid >> 5) + r * 8, sl = tid & 31;
        tile[fl][sl] = inp[((size_t)b * FF + f0 + fl) * SS + sb + sl];
    }
    __syncthreads();
    #pragma unroll
    for (int r = 0; r < 4; ++r) {
        const int tl = (tid >> 5) + r * 8, fl = tid & 31;
        xbf[(size_t)(s0 + tl) * FF + f0 + fl] = f2bf(tile[fl][tl]);
    }
}

// =====================================================================
// 0b) weight pack: f32 -> bf16 in per-lane WMMA B-fragment layout:
//     out[(((e*(N/16)+nt)*(K/32)+kc)*32 + lane)*16 + j]
//     lane: n = nt*16 + (lane&15), k = kc*32 + (lane>>4)*16 + j
//     mode 0: w is [E][K][N]; mode 1: conv w1[O=n][I][3], k = tap*1024+i
// =====================================================================
__global__ __launch_bounds__(256) void packw_kernel(
    const float* __restrict__ w, __bf16* __restrict__ out,
    int K, int N, int mode)
{
    const int gid = blockIdx.x * 256 + threadIdx.x;
    const int lane = gid & 31;
    int rest = gid >> 5;
    const int kdiv = K >> 5, ndiv = N >> 4;
    const int kc = rest % kdiv; rest /= kdiv;
    const int nt = rest % ndiv;
    const int e  = rest / ndiv;
    const int n = nt * 16 + (lane & 15);
    const int kb = kc * 32 + ((lane >> 4) << 4);
    unsigned vals[8];
    #pragma unroll
    for (int jj = 0; jj < 8; ++jj) {
        const int k0 = kb + 2 * jj, k1 = k0 + 1;
        float f0, f1;
        if (mode == 0) {
            const float* wp = w + ((size_t)e * K) * N + n;
            f0 = wp[(size_t)k0 * N];
            f1 = wp[(size_t)k1 * N];
        } else {                                   // conv: w1[(n*1024+i)*3+tap]
            f0 = w[((size_t)n * INTER + (k0 & (INTER - 1))) * 3 + (k0 >> 10)];
            f1 = w[((size_t)n * INTER + (k1 & (INTER - 1))) * 3 + (k1 >> 10)];
        }
        vals[jj] = (unsigned)f2bf_bits(f0) | ((unsigned)f2bf_bits(f1) << 16);
    }
    v4u* po = (v4u*)(out + (size_t)gid * 16);      // 32B aligned
    v4u lo = {vals[0], vals[1], vals[2], vals[3]};
    v4u hi = {vals[4], vals[5], vals[6], vals[7]};
    po[0] = lo;
    po[1] = hi;
}

// =====================================================================
// 0c) zero the 2 leading pad rows of each batch in y1pad [B][S+2][1024]
// =====================================================================
__global__ __launch_bounds__(256) void zeropad_kernel(__bf16* __restrict__ ypad)
{
    const int b = blockIdx.x;
    unsigned* p = (unsigned*)(ypad + (size_t)b * SPAD * INTER);
    for (int i = threadIdx.x; i < INTER; i += 256) p[i] = 0u;  // 2 rows = 2048 bf16
}

// =====================================================================
// 1) gate logits + softmax.  block = 256 = 16 tokens x 16 experts
// =====================================================================
__global__ __launch_bounds__(256) void gates_kernel(
    const float* __restrict__ act, const float* __restrict__ gate,
    float* __restrict__ gates, int K, int amode)
{
    __shared__ float lg[16][17];
    const int tid = threadIdx.x;
    const int e = tid & 15, tl = tid >> 4;
    const int t = blockIdx.x * 16 + tl;
    float acc = 0.f;
    if (amode == 0) {
        const int b = t >> 11, s = t & (SS - 1);
        const float* xp = act + (size_t)b * K * SS + s;
        for (int f = 0; f < K; ++f) acc += xp[(size_t)f * SS] * gate[f * NEXP + e];
    } else {
        const float* xp = act + (size_t)t * K;
        for (int f = 0; f < K; ++f) acc += xp[f] * gate[f * NEXP + e];
    }
    lg[tl][e] = acc;
    __syncthreads();
    float mx = -INFINITY;
    #pragma unroll
    for (int i = 0; i < NEXP; ++i) mx = fmaxf(mx, lg[tl][i]);
    float sum = 0.f;
    #pragma unroll
    for (int i = 0; i < NEXP; ++i) sum += expf(lg[tl][i] - mx);
    gates[(size_t)t * NEXP + e] = expf(acc - mx) / sum;
}

// =====================================================================
// 2) per-expert bitonic argsort (descending gate, ties -> lower index)
// =====================================================================
__global__ __launch_bounds__(1024) void sort_kernel(
    const float* __restrict__ gates, int* __restrict__ sortidx)
{
    extern __shared__ unsigned long long keys[];
    const int e = blockIdx.x, tid = threadIdx.x;
    for (int i = tid; i < TOK; i += 1024) {
        unsigned u = __float_as_uint(gates[(size_t)i * NEXP + e]);
        u = (u & 0x80000000u) ? ~u : (u | 0x80000000u);
        keys[i] = ((unsigned long long)u << 32) | (unsigned)(~(unsigned)i);
    }
    __syncthreads();
    for (unsigned k = 2; k <= TOK; k <<= 1) {
        for (unsigned j = k >> 1; j > 0; j >>= 1) {
            for (unsigned i = tid; i < TOK; i += 1024) {
                const unsigned p = i ^ j;
                if (p > i) {
                    const bool desc = ((i & k) == 0);
                    unsigned long long a = keys[i], b = keys[p];
                    if (desc ? (a < b) : (a > b)) { keys[i] = b; keys[p] = a; }
                }
            }
            __syncthreads();
        }
    }
    for (int i = tid; i < TOK; i += 1024)
        sortidx[e * TOK + i] = (int)(~(unsigned)keys[i]);
}

// =====================================================================
// 3) sequential capacity selection (prefix-scan ranks), reproduces the
//    reference's expert_index = inv // capacity gate-gather quirk.
// =====================================================================
__global__ __launch_bounds__(1024) void route_kernel(
    const int* __restrict__ sortidx, const float* __restrict__ gates,
    int* __restrict__ perm, float* __restrict__ rowscale)
{
    __shared__ unsigned maskw[TOK / 32];
    __shared__ int wsum[32];
    const int tid = threadIdx.x;
    const int lane = tid & 31, w = tid >> 5;
    if (tid < TOK / 32) maskw[tid] = 0u;
    __syncthreads();
    for (int e = 0; e < NEXP; ++e) {
        int tok[8], flg[8], cnt = 0;
        const int base = e * TOK + tid * 8;
        #pragma unroll
        for (int r = 0; r < 8; ++r) {
            const int t = sortidx[base + r];
            tok[r] = t;
            flg[r] = ((maskw[t >> 5] >> (t & 31)) & 1u) ? 0 : 1;
            cnt += flg[r];
        }
        int inc = cnt;
        #pragma unroll
        for (int off = 1; off < 32; off <<= 1) {
            int v = __shfl_up(inc, off, 32);
            if (lane >= off) inc += v;
        }
        if (lane == 31) wsum[w] = inc;
        __syncthreads();
        if (tid < 32) {
            int v = wsum[tid], i2 = v;
            #pragma unroll
            for (int off = 1; off < 32; off <<= 1) {
                int u = __shfl_up(i2, off, 32);
                if (tid >= off) i2 += u;
            }
            wsum[tid] = i2 - v;
        }
        __syncthreads();
        int rank = wsum[w] + (inc - cnt);
        #pragma unroll
        for (int r = 0; r < 8; ++r) {
            if (flg[r]) {
                if (rank < CAP) {
                    const int p = rank * NEXP + e;
                    const int t = tok[r];
                    perm[p] = t;
                    rowscale[p] = gates[(size_t)t * NEXP + (p >> 9)];
                    atomicOr(&maskw[t >> 5], 1u << (t & 31));
                }
                rank++;
            }
        }
        __syncthreads();
    }
}

// =====================================================================
// 4) bf16 WMMA GEMM, all-bf16 operands, branch-free, software-pipelined
//    (double-buffered fragments; K/32 is even for all call sites).
//    Wave tile 32(M) x 32(N): per 32-K chunk 8x global_load_b128 + 4 wmma.
//    A rows: amode 0 -> token-major [T,K]; amode 1 -> per-batch padded
//            rows [B][S+2][1024] (causal conv: 3 shifted windows are
//            CONTIGUOUS, tap term cancels against row stride).
//    B: pre-packed per-lane fragments (see packw_kernel).
//    perm!=0: row gather perm[row*16+e], scale by rowscale at store.
//    omode 0: Out[t*N+n] (f32 token-major); 1: Out[(b*N+n)*S+s]
// =====================================================================
__global__ __launch_bounds__(256) void wmma_gemm_kernel(
    const __bf16* __restrict__ A, const __bf16* __restrict__ Bpk,
    float* __restrict__ Out, const int* __restrict__ perm,
    const float* __restrict__ rowscale, int K, int N, int amode, int omode)
{
    const int lane = threadIdx.x & 31;
    const int m16 = lane & 15, hi = lane >> 4;
    const int e = blockIdx.z;
    const int mbase = blockIdx.y * 32;
    const int nt0 = (blockIdx.x * 8 + threadIdx.y) * 2;   // ntile pair

    const __bf16* apb[2];
    #pragma unroll
    for (int mi = 0; mi < 2; ++mi) {
        const int row = mbase + mi * 16 + m16;
        const int t = perm ? perm[row * NEXP + e] : row;
        const size_t abase = (amode == 0)
            ? (size_t)t * K
            : ((size_t)(t >> 11) * SPAD + (t & (SS - 1))) * INTER;
        apb[mi] = A + abase + hi * 16;
    }
    const int kdiv = K >> 5;
    const __bf16* bp0 = Bpk + (((size_t)e * (N >> 4) + nt0) * kdiv * 32 + lane) * 16;
    const __bf16* bp1 = bp0 + (size_t)kdiv * 512;

    v8f acc[2][2] = {};

    auto ldA = [&](int mi, int off) {
        return load_frag(apb[mi] + off, apb[mi] + off + 8);
    };
    auto ldB = [&](const __bf16* bp, int off) {
        const __bf16* p = bp + (size_t)off * 16;      // 512 elems per 32-K chunk
        return load_frag(p, p + 8);
    };
    auto mm4 = [&](v16bf (&af)[2], v16bf (&bf)[2]) {
        #pragma unroll
        for (int mi = 0; mi < 2; ++mi)
            #pragma unroll
            for (int ni = 0; ni < 2; ++ni)
                acc[mi][ni] = __builtin_amdgcn_wmma_f32_16x16x32_bf16(
                    false, af[mi], false, bf[ni], (short)0, acc[mi][ni],
                    false, false);
    };

    v16bf a0[2], b0[2], a1[2], b1[2];
    a0[0] = ldA(0, 0); a0[1] = ldA(1, 0);
    b0[0] = ldB(bp0, 0); b0[1] = ldB(bp1, 0);

    int kk = 0;
    for (; kk + 64 < K; kk += 64) {                    // K is a multiple of 64
        a1[0] = ldA(0, kk + 32); a1[1] = ldA(1, kk + 32);
        b1[0] = ldB(bp0, kk + 32); b1[1] = ldB(bp1, kk + 32);
        mm4(a0, b0);
        a0[0] = ldA(0, kk + 64); a0[1] = ldA(1, kk + 64);
        b0[0] = ldB(bp0, kk + 64); b0[1] = ldB(bp1, kk + 64);
        mm4(a1, b1);
    }
    a1[0] = ldA(0, kk + 32); a1[1] = ldA(1, kk + 32);
    b1[0] = ldB(bp0, kk + 32); b1[1] = ldB(bp1, kk + 32);
    mm4(a0, b0);
    mm4(a1, b1);

    #pragma unroll
    for (int mi = 0; mi < 2; ++mi) {
        #pragma unroll
        for (int r = 0; r < 8; ++r) {
            const int m = mbase + mi * 16 + hi * 8 + r;
            int t; float sc = 1.0f;
            if (perm) { const int p = m * NEXP + e; t = perm[p]; sc = rowscale[p]; }
            else t = m;
            #pragma unroll
            for (int ni = 0; ni < 2; ++ni) {
                const float val = acc[mi][ni][r] * sc;
                const int nf = (nt0 + ni) * 16 + m16;
                if (omode == 0) {
                    Out[(size_t)t * N + nf] = val;
                } else {
                    const int b = t >> 11, s = t & (SS - 1);
                    Out[((size_t)b * N + nf) * SS + s] = val;
                }
            }
        }
    }
}

// =====================================================================
// 5) triple_norm: [T,3072] f32 -> bf16 (optionally row-padded) + f32
// =====================================================================
__device__ __forceinline__ float block_reduce_256(float v) {
    __shared__ float red[8];
    const int lane = threadIdx.x & 31, w = threadIdx.x >> 5;
    #pragma unroll
    for (int off = 16; off > 0; off >>= 1) v += __shfl_down(v, off, 32);
    if (lane == 0) red[w] = v;
    __syncthreads();
    if (w == 0) {
        float r = (lane < 8) ? red[lane] : 0.f;
        #pragma unroll
        for (int off = 4; off > 0; off >>= 1) r += __shfl_down(r, off, 32);
        if (lane == 0) red[0] = r;
    }
    __syncthreads();
    const float r = red[0];
    __syncthreads();
    return r;
}

__global__ __launch_bounds__(256) void norm_kernel(
    const float* __restrict__ X, __bf16* __restrict__ Ybf,
    float* __restrict__ Yf, int opad)
{
    const int t = blockIdx.x, tid = threadIdx.x;
    const float* xp = X + (size_t)t * N3I;
    float y[4], sum = 0.f;
    #pragma unroll
    for (int r = 0; r < 4; ++r) {
        const int c = tid + r * 256;
        const float s0 = xp[c], s1 = xp[INTER + c], sh = xp[2 * INTER + c];
        const float rr = fmaxf(s0, 0.f);
        y[r] = rr * rr * rr * s1 + sh;
        sum += y[r];
    }
    const float mean = block_reduce_256(sum) * (1.0f / (float)INTER);
    float ss = 0.f;
    #pragma unroll
    for (int r = 0; r < 4; ++r) { y[r] -= mean; ss += y[r] * y[r]; }
    const float rstd = 32.0f / sqrtf(block_reduce_256(ss));   // 1024^(1/2)=32
    const size_t orow = opad
        ? ((size_t)(t >> 11) * SPAD + (t & (SS - 1)) + 2)
        : (size_t)t;
    #pragma unroll
    for (int r = 0; r < 4; ++r) {
        const int c = tid + r * 256;
        const float v = y[r] * rstd;
        Ybf[orow * INTER + c] = f2bf(v);
        if (Yf) Yf[(size_t)t * INTER + c] = v;
    }
}

// =====================================================================
// host launcher
// =====================================================================
extern "C" void kernel_launch(void* const* d_in, const int* in_sizes, int n_in,
                              void* d_out, int out_size, void* d_ws, size_t ws_size,
                              hipStream_t stream)
{
    const float* inp   = (const float*)d_in[0];
    const float* w0    = (const float*)d_in[1];
    const float* gate0 = (const float*)d_in[2];
    const float* w1    = (const float*)d_in[3];
    const float* w2    = (const float*)d_in[4];
    const float* gate2 = (const float*)d_in[5];
    float* out = (float*)d_out;

    char* ws = (char*)d_ws;
    size_t off = 0;
    auto take = [&](size_t bytes) -> void* {
        off = (off + 255) & ~(size_t)255;
        void* p = ws + off; off += bytes; return p;
    };
    float*  bufA     = (float*) take(sizeof(float)  * (size_t)TOK * N3I);  // out1 / conv out
    float*  y2f      = (float*) take(sizeof(float)  * (size_t)TOK * INTER);
    float*  gates    = (float*) take(sizeof(float)  * (size_t)TOK * NEXP);
    __bf16* xbf      = (__bf16*)take(sizeof(__bf16) * (size_t)TOK * FF);
    __bf16* y1pad    = (__bf16*)take(sizeof(__bf16) * (size_t)BB * SPAD * INTER);
    __bf16* y2bf     = (__bf16*)take(sizeof(__bf16) * (size_t)TOK * INTER);
    __bf16* w0pk     = (__bf16*)take(sizeof(__bf16) * (size_t)NEXP * FF * N3I);
    __bf16* cwpk     = (__bf16*)take(sizeof(__bf16) * (size_t)N3I * N3I);
    __bf16* w2pk     = (__bf16*)take(sizeof(__bf16) * (size_t)NEXP * INTER * FF);
    int*    sortidx  = (int*)   take(sizeof(int)    * (size_t)NEXP * TOK);
    int*    perm     = (int*)   take(sizeof(int)    * TOK);
    float*  rowscale = (float*) take(sizeof(float)  * TOK);
    (void)ws_size; (void)in_sizes; (void)n_in; (void)out_size;

    const dim3 wgBlk(32, 8);

    // ---- one-time packs / transposes (per launch, deterministic) ----
    tcvt_kernel<<<dim3(TOK / 32, FF / 32), 256, 0, stream>>>(inp, xbf);
    packw_kernel<<<(NEXP * (N3I / 16) * (FF / 32) * 32) / 256, 256, 0, stream>>>(
        w0, w0pk, FF, N3I, 0);
    packw_kernel<<<((N3I / 16) * (N3I / 32) * 32) / 256, 256, 0, stream>>>(
        w1, cwpk, N3I, N3I, 1);
    packw_kernel<<<(NEXP * (FF / 16) * (INTER / 32) * 32) / 256, 256, 0, stream>>>(
        w2, w2pk, INTER, FF, 0);
    zeropad_kernel<<<BB, 256, 0, stream>>>(y1pad);

    // ---- MoE 1 ----
    gates_kernel<<<TOK / 16, 256, 0, stream>>>(inp, gate0, gates, FF, 0);
    sort_kernel<<<NEXP, 1024, TOK * sizeof(unsigned long long), stream>>>(gates, sortidx);
    route_kernel<<<1, 1024, 0, stream>>>(sortidx, gates, perm, rowscale);
    wmma_gemm_kernel<<<dim3(N3I / 256, CAP / 32, NEXP), wgBlk, 0, stream>>>(
        xbf, w0pk, bufA, perm, rowscale, FF, N3I, /*amode*/0, /*omode*/0);

    // ---- norm 1 (bf16 into padded rows; feeds conv GEMM) ----
    norm_kernel<<<TOK, 256, 0, stream>>>(bufA, y1pad, nullptr, /*opad*/1);

    // ---- causal conv as GEMM (K = 3 taps x 1024, contiguous via padding) ----
    wmma_gemm_kernel<<<dim3(N3I / 256, TOK / 32, 1), wgBlk, 0, stream>>>(
        y1pad, cwpk, bufA, nullptr, nullptr, N3I, N3I, /*amode*/1, /*omode*/0);

    // ---- norm 2 (bf16 for GEMM + f32 for gating) ----
    norm_kernel<<<TOK, 256, 0, stream>>>(bufA, y2bf, y2f, /*opad*/0);

    // ---- MoE 2 (writes final [B,F,S] output) ----
    gates_kernel<<<TOK / 16, 256, 0, stream>>>(y2f, gate2, gates, INTER, 1);
    sort_kernel<<<NEXP, 1024, TOK * sizeof(unsigned long long), stream>>>(gates, sortidx);
    route_kernel<<<1, 1024, 0, stream>>>(sortidx, gates, perm, rowscale);
    wmma_gemm_kernel<<<dim3(FF / 256, CAP / 32, NEXP), wgBlk, 0, stream>>>(
        y2bf, w2pk, out, perm, rowscale, INTER, FF, /*amode*/0, /*omode*/1);
}
